// DualMem_49357764165819
// MI455X (gfx1250) — matrix operand order repaired and checked by reading, not compiled, not run on hardware
//
#include <hip/hip_runtime.h>

// ---------------------------------------------------------------------------
// DualMem fast_get_image_pred via the Gram identity:
//   logits[b,c] = 100 * (w.sim) / sqrt(w^T G_c w),  w = exp(-beta(1-sim))
// Streams mem[1000,11,1024] exactly once; never materializes adapt [B,C,D].
// bf16 WMMA (f32 accum) for sim and Gram tiles. Wave32 / gfx1250.
//
// v3: fully-unrolled WMMA loops with explicit double-buffered fragments
//     (kills v_mov_b64 rotation copies + WAR-hazard v_nops seen in v2),
//     Gram partials merged via ds_add_f32 atomics into one tile
//     (4x less epilogue LDS/VALU), 3 barriers/class, 2 WGs/WGP kept.
// ---------------------------------------------------------------------------

#define BETA 5.5f

typedef __bf16        v16bf __attribute__((ext_vector_type(16)));
typedef float         v8f   __attribute__((ext_vector_type(8)));
typedef unsigned int  u32x4 __attribute__((ext_vector_type(4)));
typedef unsigned int  u32x2 __attribute__((ext_vector_type(2)));
typedef float         f32x4 __attribute__((ext_vector_type(4)));

constexpr int kB = 64;      // batch
constexpr int kC = 1000;    // classes
constexpr int kM = 11;      // memories per class
constexpr int kD = 1024;    // feature dim

constexpr int STRIDE_E = kD + 8;         // bf16 elems per LDS row (pad -> conflict-free b128)
constexpr int STRIDE_B = STRIDE_E * 2;   // 2064 bytes, 16B-aligned

constexpr int IMG_OFF  = 0;
constexpr int MEM_OFF  = IMG_OFF + kB * STRIDE_B;    // 132096
constexpr int SIM_OFF  = MEM_OFF + kM * STRIDE_B;    // 154800
constexpr int GRAM_OFF = SIM_OFF + kB * 16 * 4;      // 158896
constexpr int SMEM_BYTES = GRAM_OFF + 16 * 16 * 4;   // 159920 (< 160 KB -> 2 WG/WGP)

__device__ inline unsigned pack2_bf16(float a, float b) {
  unsigned ua = __builtin_bit_cast(unsigned, a);
  unsigned ub = __builtin_bit_cast(unsigned, b);
  ua = (ua + 0x7FFFu + ((ua >> 16) & 1u)) >> 16;   // round-to-nearest-even
  ub = (ub + 0x7FFFu + ((ub >> 16) & 1u)) >> 16;
  return (ua & 0xFFFFu) | (ub << 16);
}

__global__ __launch_bounds__(128)
void dualmem_kernel(const float* __restrict__ img,
                    const float* __restrict__ mem,
                    float* __restrict__ out,
                    int nclass) {
  extern __shared__ __align__(16) char smem[];
  char*  img_lds  = smem + IMG_OFF;
  char*  mem_lds  = smem + MEM_OFF;
  float* sim_lds  = (float*)(smem + SIM_OFF);
  float* gram_lds = (float*)(smem + GRAM_OFF);   // single 16x16 tile

  const int tid  = threadIdx.x;
  const int wid  = tid >> 5;        // wave id (wave32)
  const int lane = tid & 31;
  const int g    = lane >> 4;       // half-wave group
  const int r    = lane & 15;       // row/col within tile

  // ---- once per WG: img f32 -> bf16 into LDS (padded rows) ----
  {
    const f32x4* img4 = (const f32x4*)img;         // 16384 float4
    #pragma unroll 4
    for (int i = 0; i < 128; i++) {
      int q    = tid + i * 128;
      int row  = q >> 8;            // 256 float4 per row
      int colv = q & 255;
      f32x4 v = img4[q];
      u32x2 p;
      p[0] = pack2_bf16(v[0], v[1]);
      p[1] = pack2_bf16(v[2], v[3]);
      *(u32x2*)(img_lds + row * STRIDE_B + colv * 8) = p;
    }
  }
  __syncthreads();

  // Per-lane fragment row pointers (ISA 16-bit A/B VGPR layouts).
  const u32x4* rowA  = (const u32x4*)(img_lds + (16 * wid + r) * STRIDE_B);
  const int    nr    = (r < kM) ? r : (kM - 1);    // clamp: rows >= 11 masked later
  const u32x4* rowB  = (const u32x4*)(mem_lds + nr * STRIDE_B);
  const u32x4* rowGA = rowB;                       // Gram A rows = mem rows

  for (int c = blockIdx.x; c < nclass; c += gridDim.x) {
    // ---- load mem[c] f32 -> bf16 into LDS; zero Gram tile ----
    {
      const f32x4* mem4 = (const f32x4*)(mem + (size_t)c * (kM * kD));  // 2816 float4
      #pragma unroll 2
      for (int i = 0; i < 22; i++) {
        int q    = tid + i * 128;
        int row  = q >> 8;
        int colv = q & 255;
        f32x4 v = mem4[q];
        u32x2 p;
        p[0] = pack2_bf16(v[0], v[1]);
        p[1] = pack2_bf16(v[2], v[3]);
        *(u32x2*)(mem_lds + row * STRIDE_B + colv * 8) = p;
      }
      gram_lds[tid]       = 0.f;
      gram_lds[tid + 128] = 0.f;
    }
    // Prefetch next class's tile into cache (global_prefetch_b8; no counters).
    {
      int cn = c + gridDim.x;
      if (cn < nclass) {
        const char* nxt = (const char*)(mem + (size_t)cn * (kM * kD));
        #pragma unroll
        for (int i = 0; i < 3; i++) {
          int off = (tid + i * 128) * 128;         // 128B cachelines
          if (off < kM * kD * 4) __builtin_prefetch(nxt + off, 0, 2);
        }
      }
    }
    __syncthreads();   // B1: tile + zeroed gram visible

    // ---- WMMA sim tile per wave (16 batch rows x 16 mem cols) ----
    // Fully unrolled, explicit double-buffered fragments: loads run one
    // k-step ahead (partial dscnt waits), no loop-carried register rotation.
    v8f acc = {0.f, 0.f, 0.f, 0.f, 0.f, 0.f, 0.f, 0.f};
    {
      u32x4 a0[2], a1[2], b0[2], b1[2];
      a0[0] = rowA[g];     a1[0] = rowA[2 + g];
      b0[0] = rowB[g]; b1[0] = rowB[2 + g];
      #pragma unroll
      for (int ks = 0; ks < 32; ks++) {
        const int cur = ks & 1, nxt = cur ^ 1;
        if (ks < 31) {
          a0[nxt] = rowA[4 * (ks + 1) + g];
          a1[nxt] = rowA[4 * (ks + 1) + 2 + g];
          b0[nxt] = rowB[4 * (ks + 1) + g];
          b1[nxt] = rowB[4 * (ks + 1) + 2 + g];
        }
        union { u32x4 u[2]; v16bf v; } fa, fb;
        fa.u[0] = a0[cur]; fa.u[1] = a1[cur];
        fb.u[0] = b0[cur]; fb.u[1] = b1[cur];
        acc = __builtin_amdgcn_wmma_f32_16x16x32_bf16(
            false, fa.v, false, fb.v, (short)0, acc, false, false);
      }
    }

    // ---- Gram partial per wave: wave w accumulates k-steps [8w, 8w+8) ----
    v8f accG = {0.f, 0.f, 0.f, 0.f, 0.f, 0.f, 0.f, 0.f};
    {
      const int ks0 = 8 * wid;
      u32x4 ga0[2], ga1[2], gb0[2], gb1[2];
      ga0[0] = rowGA[4 * ks0 + g];    ga1[0] = rowGA[4 * ks0 + 2 + g];
      gb0[0] = rowB[4 * ks0 + g]; gb1[0] = rowB[4 * ks0 + 2 + g];
      #pragma unroll
      for (int j = 0; j < 8; j++) {
        const int cur = j & 1, nxt = cur ^ 1;
        if (j < 7) {
          int ks = ks0 + j + 1;
          ga0[nxt] = rowGA[4 * ks + g];
          ga1[nxt] = rowGA[4 * ks + 2 + g];
          gb0[nxt] = rowB[4 * ks + g];
          gb1[nxt] = rowB[4 * ks + 2 + g];
        }
        union { u32x4 u[2]; v16bf v; } fg, fb;
        fg.u[0] = ga0[cur]; fg.u[1] = ga1[cur];
        fb.u[0] = gb0[cur]; fb.u[1] = gb1[cur];
        accG = __builtin_amdgcn_wmma_f32_16x16x32_bf16(
            false, fg.v, false, fb.v, (short)0, accG, false, false);
      }
    }

    // C/D layout: VGPR i -> row M=i(+8g), col N = lane&15
    #pragma unroll
    for (int i = 0; i < 8; i++)
      sim_lds[(16 * wid + i + 8 * g) * 16 + r] = acc[i];
    // Merge Gram partials across waves with LDS float atomics (ds_add_f32).
    #pragma unroll
    for (int i = 0; i < 8; i++)
      atomicAdd(&gram_lds[(i + 8 * g) * 16 + r], accG[i]);
    __syncthreads();   // B2: sim + merged gram visible

    // ---- epilogue: logits[b,c] = 100 * (w.sim) / sqrt(w^T G w) ----
    if (tid < kB) {
      float s[kM], w[kM];
      float numer = 0.f;
      #pragma unroll
      for (int m = 0; m < kM; m++) {
        s[m] = sim_lds[tid * 16 + m];
        w[m] = __expf(BETA * (s[m] - 1.f));
        numer += w[m] * s[m];
      }
      float den2 = 0.f;
      #pragma unroll
      for (int m = 0; m < kM; m++) {
        float t = 0.f;
        #pragma unroll
        for (int m2 = 0; m2 < kM; m2++) t += gram_lds[m * 16 + m2] * w[m2];
        den2 += w[m] * t;
      }
      out[(size_t)tid * nclass + c] = 100.f * numer * rsqrtf(den2);
    }
    __syncthreads();   // B3: fence epilogue reads vs next iter's gram zeroing
  }
}

extern "C" void kernel_launch(void* const* d_in, const int* in_sizes, int n_in,
                              void* d_out, int out_size, void* d_ws, size_t ws_size,
                              hipStream_t stream) {
  (void)in_sizes; (void)n_in; (void)d_ws; (void)ws_size; (void)out_size;
  const float* img = (const float*)d_in[0];   // [64, 1024] f32, unit-norm rows
  const float* mem = (const float*)d_in[1];   // [1000, 11, 1024] f32
  float* out = (float*)d_out;                 // [64, 1000] f32

  hipFuncSetAttribute(reinterpret_cast<const void*>(dualmem_kernel),
                      hipFuncAttributeMaxDynamicSharedMemorySize, SMEM_BYTES);
  dim3 grid(250), block(128);   // grid-stride: 4 classes per WG
  dualmem_kernel<<<grid, block, SMEM_BYTES, stream>>>(img, mem, out, kC);
}
